// Feedback_Code_46823733461556
// MI455X (gfx1250) — matrix operand order repaired and checked by reading, not compiled, not run on hardware
//
#include <hip/hip_runtime.h>
#include <hip/hip_bf16.h>
#include <math.h>

#define BATCH 8192
#define TT    18
#define NBITS 6
#define HID   256
#define G3    768   // 3 * HID gate width

typedef __attribute__((ext_vector_type(16))) _Float16 v16h;
typedef __attribute__((ext_vector_type(8)))  _Float16 v8h;
typedef __attribute__((ext_vector_type(8)))  float    v8f;

__device__ __forceinline__ float sigmoidf_(float x) { return 1.f / (1.f + __expf(-x)); }

// CDNA5 async global->LDS copy, 16B per lane, tracked by ASYNCcnt.
__device__ __forceinline__ void async_copy_b128(void* lds_dst, const void* gsrc) {
  unsigned l = (unsigned)(uintptr_t)lds_dst;       // low 32 bits = LDS offset
  unsigned long long g = (unsigned long long)(uintptr_t)gsrc;
  asm volatile("global_load_async_to_lds_b128 %0, %1, off" :: "v"(l), "v"(g) : "memory");
}
__device__ __forceinline__ void wait_async0() {
  asm volatile("s_wait_asynccnt 0x0" ::: "memory");
}

// pack two contiguous v8h LDS loads into one WMMA fragment
__device__ __forceinline__ v16h frag16(const _Float16* p, int gap) {
  v8h a0 = *(const v8h*)p;
  v8h a1 = *(const v8h*)(p + gap);
  v16h r;
#pragma unroll
  for (int i = 0; i < 8; ++i) { r[i] = a0[i]; r[i + 8] = a1[i]; }
  return r;
}

// ---------------------------------------------------------------------------
// Tiled WMMA GEMM:  G[B][768] = A[B][K] * W[768][K]^T
// A: f16 row-major [B][K].  Bw: f16 row-major [768][K] (original W layout).
// Block: 256 threads (8 wave32), 128x64 output tile, K in 64-chunks,
// double-buffered LDS filled by GLOBAL_LOAD_ASYNC_TO_LDS_B128.
// Each wave owns a 2x2 grid of 16x16 tiles -> 4 WMMA per k-step.
// ---------------------------------------------------------------------------
template <int K>
__global__ void __launch_bounds__(256) gemm_kernel(const _Float16* __restrict__ A,
                                                   const _Float16* __restrict__ Bw,
                                                   float* __restrict__ G) {
  __shared__ __align__(16) _Float16 As[2][128][72];  // [buf][m][k]
  __shared__ __align__(16) _Float16 Bs[2][64][72];   // [buf][n][k]
  const int tid  = threadIdx.x;
  const int m0   = blockIdx.x * 128;
  const int n0   = blockIdx.y * 64;
  const int w    = tid >> 5;
  const int lane = tid & 31;
  const int mt0  = (w & 3) << 1;       // two M tiles: mt0, mt0+1
  const int nt0  = (w >> 2) << 1;      // two N tiles: nt0, nt0+1
  const int hi   = lane >> 4;
  const int lm   = lane & 15;

  v8f c00 = {}, c01 = {}, c10 = {}, c11 = {};

  auto stage = [&](int buf, int kc) {
    // A tile: 128 rows x 128B  -> 4 async b128 per thread
#pragma unroll
    for (int it = 0; it < 4; ++it) {
      int c = tid + 256 * it;
      int row = c >> 3, cb = (c & 7) * 16;
      async_copy_b128((char*)&As[buf][row][0] + cb,
                      (const char*)(A + (size_t)(m0 + row) * K + kc) + cb);
    }
    // B tile: 64 rows x 128B -> 2 async b128 per thread
#pragma unroll
    for (int it = 0; it < 2; ++it) {
      int c = tid + 256 * it;
      int row = c >> 3, cb = (c & 7) * 16;
      async_copy_b128((char*)&Bs[buf][row][0] + cb,
                      (const char*)(Bw + (size_t)(n0 + row) * K + kc) + cb);
    }
  };

  stage(0, 0);
  wait_async0();
  __syncthreads();

  int buf = 0;
  for (int kc = 0; kc < K; kc += 64) {
    if (kc + 64 < K) stage(buf ^ 1, kc + 64);           // async prefill next buffer
    if (kc + 128 < K) {                                  // L2/WGP prefetch of tile after next
      __builtin_prefetch((const char*)(A + (size_t)(m0 + (tid >> 1)) * K + (kc + 128)), 0, 0);
      if (tid < 64)
        __builtin_prefetch((const char*)(Bw + (size_t)(n0 + tid) * K + (kc + 128)), 0, 0);
    }
#pragma unroll
    for (int ks = 0; ks < 64; ks += 32) {
      v16h a0 = frag16(&As[buf][mt0 * 16 + lm][ks + hi * 8], 16);
      v16h a1 = frag16(&As[buf][(mt0 + 1) * 16 + lm][ks + hi * 8], 16);
      v16h b0 = frag16(&Bs[buf][nt0 * 16 + lm][ks + hi * 8], 16);
      v16h b1 = frag16(&Bs[buf][(nt0 + 1) * 16 + lm][ks + hi * 8], 16);
      c00 = __builtin_amdgcn_wmma_f32_16x16x32_f16(false, a0, false, b0, (short)0, c00, false, false);
      c01 = __builtin_amdgcn_wmma_f32_16x16x32_f16(false, a0, false, b1, (short)0, c01, false, false);
      c10 = __builtin_amdgcn_wmma_f32_16x16x32_f16(false, a1, false, b0, (short)0, c10, false, false);
      c11 = __builtin_amdgcn_wmma_f32_16x16x32_f16(false, a1, false, b1, (short)0, c11, false, false);
    }
    wait_async0();        // this wave's fills of the next buffer are done
    __syncthreads();      // everyone finished reading current buffer
    buf ^= 1;
  }

  auto store_tile = [&](v8f c, int mt, int nt) {
#pragma unroll
    for (int r = 0; r < 8; ++r) {
      int row = m0 + mt * 16 + r + hi * 8;
      G[(size_t)row * G3 + n0 + nt * 16 + lm] = c[r];
    }
  };
  store_tile(c00, mt0, nt0);
  store_tile(c01, mt0, nt0 + 1);
  store_tile(c10, mt0 + 1, nt0);
  store_tile(c11, mt0 + 1, nt0 + 1);
}

// ---------------------------------------------------------------------------
// straight f32 -> f16 copy (weights keep original [768][K] row-major layout)
// ---------------------------------------------------------------------------
__global__ void convert_f16(const float* __restrict__ W, _Float16* __restrict__ Wt, int count) {
  int idx = blockIdx.x * 256 + threadIdx.x;
  if (idx < count) Wt[idx] = (_Float16)W[idx];
}

__global__ void zero_f32(float* p, size_t n) {
  size_t i = (size_t)blockIdx.x * blockDim.x + threadIdx.x;
  size_t stride = (size_t)gridDim.x * blockDim.x;
  for (; i < n; i += stride) p[i] = 0.f;
}

// normalize power/merge weights: w' = sqrt(w^2 * T / sum(w^2))
__global__ void prep_small(const float* __restrict__ weight_power,
                           const float* __restrict__ weight_merge,
                           float* wp, float* wf, float* wb) {
  __shared__ float s[3];
  if (threadIdx.x == 0) {
    float a = 0.f, bm = 0.f, c = 0.f;
    for (int t = 0; t < TT; ++t) {
      float p = weight_power[t];
      float f = weight_merge[2 * t], bk = weight_merge[2 * t + 1];
      a += p * p; bm += f * f; c += bk * bk;
    }
    s[0] = a; s[1] = bm; s[2] = c;
  }
  __syncthreads();
  int t = threadIdx.x;
  if (t < TT) {
    float p = weight_power[t];
    float f = weight_merge[2 * t], bk = weight_merge[2 * t + 1];
    wp[t] = sqrtf(p * p * (float)TT / s[0]);
    wf[t] = sqrtf(f * f * (float)TT / s[1]);
    wb[t] = sqrtf(bk * bk * (float)TT / s[2]);
  }
}

// ---------------------------------------------------------------------------
// Encoder layer 0 GRU combine: gi computed directly (K=7), gh from GEMM buffer
// writes f32 state + f16 shadow (next GEMM's A operand)
// ---------------------------------------------------------------------------
__global__ void __launch_bounds__(256) enc_combine0(const float* __restrict__ bbits,
                                                    const float* __restrict__ z,
                                                    const float* __restrict__ Wih,
                                                    const float* __restrict__ bih,
                                                    const float* __restrict__ bhh,
                                                    const float* __restrict__ Ggh,
                                                    float* __restrict__ h,
                                                    _Float16* __restrict__ h16) {
  int idx = blockIdx.x * 256 + threadIdx.x;
  int row = idx >> 8, j = idx & 255;
  float x[7];
#pragma unroll
  for (int c = 0; c < NBITS; ++c) x[c] = bbits[row * NBITS + c];
  x[6] = z[row];
  float gi[3];
#pragma unroll
  for (int g = 0; g < 3; ++g) {
    const float* wr = Wih + (size_t)(g * HID + j) * 7;
    float s = bih[g * HID + j];
#pragma unroll
    for (int c = 0; c < 7; ++c) s += x[c] * wr[c];
    gi[g] = s;
  }
  size_t gb = (size_t)row * G3 + j;
  float ghr = Ggh[gb] + bhh[j];
  float ghz = Ggh[gb + HID] + bhh[HID + j];
  float ghn = Ggh[gb + 2 * HID] + bhh[2 * HID + j];
  float r  = sigmoidf_(gi[0] + ghr);
  float zz = sigmoidf_(gi[1] + ghz);
  float n  = tanhf(gi[2] + r * ghn);
  size_t hb = (size_t)row * HID + j;
  float hn = (1.f - zz) * n + zz * h[hb];
  h[hb] = hn;
  h16[hb] = (_Float16)hn;
}

// generic GRU combine when both gi and gh come from GEMM buffers
__global__ void __launch_bounds__(256) combine2(const float* __restrict__ Ggi,
                                                const float* __restrict__ Ggh,
                                                const float* __restrict__ bih,
                                                const float* __restrict__ bhh,
                                                float* __restrict__ h,
                                                _Float16* __restrict__ h16) {
  int idx = blockIdx.x * 256 + threadIdx.x;
  int row = idx >> 8, j = idx & 255;
  size_t gb = (size_t)row * G3 + j;
  float gir = Ggi[gb] + bih[j];
  float giz = Ggi[gb + HID] + bih[HID + j];
  float gin = Ggi[gb + 2 * HID] + bih[2 * HID + j];
  float ghr = Ggh[gb] + bhh[j];
  float ghz = Ggh[gb + HID] + bhh[HID + j];
  float ghn = Ggh[gb + 2 * HID] + bhh[2 * HID + j];
  float r  = sigmoidf_(gir + ghr);
  float zz = sigmoidf_(giz + ghz);
  float n  = tanhf(gin + r * ghn);
  size_t hb = (size_t)row * HID + j;
  float hn = (1.f - zz) * n + zz * h[hb];
  h[hb] = hn;
  h16[hb] = (_Float16)hn;
}

// per-row linear + tanh, accumulate batch stats via atomics (one wave32 per row)
__global__ void __launch_bounds__(256) enc_rowreduce(const float* __restrict__ h1,
                                                     const float* __restrict__ linW,
                                                     const float* __restrict__ linb,
                                                     float* __restrict__ xt,
                                                     float* __restrict__ stats) {
  int wid  = (blockIdx.x * 256 + threadIdx.x) >> 5;
  int lane = threadIdx.x & 31;
  if (wid >= BATCH) return;
  float s = 0.f;
  for (int k = lane; k < HID; k += 32) s += h1[(size_t)wid * HID + k] * linW[k];
#pragma unroll
  for (int off = 16; off > 0; off >>= 1) s += __shfl_xor(s, off, 32);
  if (lane == 0) {
    float x = tanhf(s + linb[0]);
    xt[wid] = x;
    atomicAdd(&stats[0], x);
    atomicAdd(&stats[1], x * x);
  }
}

// finalize batchnorm, emit channel output y and feedback z
__global__ void __launch_bounds__(256) enc_emit(const float* __restrict__ xt,
                                                const float* __restrict__ stats,
                                                const float* __restrict__ wp, int t,
                                                const float* __restrict__ noise1,
                                                const float* __restrict__ noise2,
                                                float* __restrict__ y_seq,
                                                float* __restrict__ z) {
  int row = blockIdx.x * 256 + threadIdx.x;
  if (row >= BATCH) return;
  float mean   = stats[0] * (1.f / (float)BATCH);
  float var    = (stats[1] - (float)BATCH * mean * mean) * (1.f / (float)(BATCH - 1));
  float invstd = rsqrtf(var);
  float xn = (xt[row] - mean) * invstd;
  float y  = xn * wp[t] + noise1[(size_t)row * TT + t];
  y_seq[(size_t)t * BATCH + row] = y;
  z[row] = y + noise2[(size_t)row * TT + t];
}

// decoder layer 0 combine (input dim 1): writes f32 state, f16 shadow, f16 s1 slot
__global__ void __launch_bounds__(256) dec0_combine(const float* __restrict__ y,
                                                    const float* __restrict__ Wih,
                                                    const float* __restrict__ bih,
                                                    const float* __restrict__ bhh,
                                                    const float* __restrict__ Ggh,
                                                    float* __restrict__ h,
                                                    _Float16* __restrict__ h16,
                                                    _Float16* __restrict__ s1out) {
  int idx = blockIdx.x * 256 + threadIdx.x;
  int row = idx >> 8, j = idx & 255;
  float yv  = y[row];
  float gir = yv * Wih[j] + bih[j];
  float giz = yv * Wih[HID + j] + bih[HID + j];
  float gin = yv * Wih[2 * HID + j] + bih[2 * HID + j];
  size_t gb = (size_t)row * G3 + j;
  float r  = sigmoidf_(gir + Ggh[gb] + bhh[j]);
  float zz = sigmoidf_(giz + Ggh[gb + HID] + bhh[HID + j]);
  float n  = tanhf(gin + r * (Ggh[gb + 2 * HID] + bhh[2 * HID + j]));
  size_t hb = (size_t)row * HID + j;
  float hn = (1.f - zz) * n + zz * h[hb];
  h[hb] = hn;
  h16[hb] = (_Float16)hn;
  s1out[(size_t)row * (2 * HID) + j] = (_Float16)hn;
}

// decoder layer 1 combine + on-the-fly weighted time merge
__global__ void __launch_bounds__(256) dec1_combine(const float* __restrict__ Ggi,
                                                    const float* __restrict__ Ggh,
                                                    const float* __restrict__ bih,
                                                    const float* __restrict__ bhh,
                                                    float* __restrict__ h,
                                                    _Float16* __restrict__ h16,
                                                    float* __restrict__ racc,
                                                    const float* __restrict__ wvec, int t) {
  int idx = blockIdx.x * 256 + threadIdx.x;
  int row = idx >> 8, j = idx & 255;
  size_t gb = (size_t)row * G3 + j;
  float gir = Ggi[gb] + bih[j];
  float giz = Ggi[gb + HID] + bih[HID + j];
  float gin = Ggi[gb + 2 * HID] + bih[2 * HID + j];
  float r  = sigmoidf_(gir + Ggh[gb] + bhh[j]);
  float zz = sigmoidf_(giz + Ggh[gb + HID] + bhh[HID + j]);
  float n  = tanhf(gin + r * (Ggh[gb + 2 * HID] + bhh[2 * HID + j]));
  size_t hb = (size_t)row * HID + j;
  float hn = (1.f - zz) * n + zz * h[hb];
  h[hb] = hn;
  h16[hb] = (_Float16)hn;
  racc[hb] += wvec[t] * hn;
}

// final head: sigmoid([rf,rb] @ W.T + b) -> [B, 6]
__global__ void __launch_bounds__(256) final_out(const float* __restrict__ rf,
                                                 const float* __restrict__ rb,
                                                 const float* __restrict__ W,
                                                 const float* __restrict__ bias,
                                                 float* __restrict__ out) {
  int idx = blockIdx.x * 256 + threadIdx.x;
  if (idx >= BATCH * NBITS) return;
  int row = idx / NBITS, o = idx % NBITS;
  const float* wr = W + (size_t)o * (2 * HID);
  float s = bias[o];
  for (int k = 0; k < HID; ++k) s += rf[(size_t)row * HID + k] * wr[k];
  for (int k = 0; k < HID; ++k) s += rb[(size_t)row * HID + k] * wr[HID + k];
  out[idx] = sigmoidf_(s);
}

// ---------------------------------------------------------------------------
extern "C" void kernel_launch(void* const* d_in, const int* in_sizes, int n_in,
                              void* d_out, int out_size, void* d_ws, size_t ws_size,
                              hipStream_t stream) {
  (void)in_sizes; (void)n_in; (void)out_size; (void)ws_size;
  const float* bbits      = (const float*)d_in[0];
  const float* noise1     = (const float*)d_in[1];
  const float* noise2     = (const float*)d_in[2];
  const float* enc_Wih0   = (const float*)d_in[3];
  const float* enc_Whh0   = (const float*)d_in[4];
  const float* enc_bih0   = (const float*)d_in[5];
  const float* enc_bhh0   = (const float*)d_in[6];
  const float* enc_Wih1   = (const float*)d_in[7];
  const float* enc_Whh1   = (const float*)d_in[8];
  const float* enc_bih1   = (const float*)d_in[9];
  const float* enc_bhh1   = (const float*)d_in[10];
  const float* enc_lin_W  = (const float*)d_in[11];
  const float* enc_lin_b  = (const float*)d_in[12];
  const float* weight_power = (const float*)d_in[13];
  const float* dWih0f = (const float*)d_in[14];
  const float* dWhh0f = (const float*)d_in[15];
  const float* dbih0f = (const float*)d_in[16];
  const float* dbhh0f = (const float*)d_in[17];
  const float* dWih0b = (const float*)d_in[18];
  const float* dWhh0b = (const float*)d_in[19];
  const float* dbih0b = (const float*)d_in[20];
  const float* dbhh0b = (const float*)d_in[21];
  const float* dWih1f = (const float*)d_in[22];
  const float* dWhh1f = (const float*)d_in[23];
  const float* dbih1f = (const float*)d_in[24];
  const float* dbhh1f = (const float*)d_in[25];
  const float* dWih1b = (const float*)d_in[26];
  const float* dWhh1b = (const float*)d_in[27];
  const float* dbih1b = (const float*)d_in[28];
  const float* dbhh1b = (const float*)d_in[29];
  const float* dec_lin_W  = (const float*)d_in[30];
  const float* dec_lin_b  = (const float*)d_in[31];
  const float* weight_merge = (const float*)d_in[32];
  float* out = (float*)d_out;

  char* wsp = (char*)d_ws;
  auto alloc = [&](size_t bytes) -> char* {
    char* p = wsp;
    wsp += (bytes + 255) & ~(size_t)255;
    return p;
  };

  // --- zero-initialized region (contiguous): f32 states + f16 shadows ---
  char* zstart = wsp;
  float* h0   = (float*)alloc((size_t)BATCH * HID * 4);
  float* h1   = (float*)alloc((size_t)BATCH * HID * 4);
  float* hd0f = (float*)alloc((size_t)BATCH * HID * 4);
  float* hd0b = (float*)alloc((size_t)BATCH * HID * 4);
  float* hd1f = (float*)alloc((size_t)BATCH * HID * 4);
  float* hd1b = (float*)alloc((size_t)BATCH * HID * 4);
  float* rf   = (float*)alloc((size_t)BATCH * HID * 4);
  float* rb   = (float*)alloc((size_t)BATCH * HID * 4);
  _Float16* h0_16   = (_Float16*)alloc((size_t)BATCH * HID * 2);
  _Float16* h1_16   = (_Float16*)alloc((size_t)BATCH * HID * 2);
  _Float16* hd0f16  = (_Float16*)alloc((size_t)BATCH * HID * 2);
  _Float16* hd0b16  = (_Float16*)alloc((size_t)BATCH * HID * 2);
  _Float16* hd1f16  = (_Float16*)alloc((size_t)BATCH * HID * 2);
  _Float16* hd1b16  = (_Float16*)alloc((size_t)BATCH * HID * 2);
  float* zfb  = (float*)alloc((size_t)BATCH * 4);
  float* stats = (float*)alloc((size_t)TT * 2 * 4);
  size_t zcount = (size_t)(wsp - zstart) / 4;

  // --- other scratch ---
  _Float16* wEncWhh0  = (_Float16*)alloc((size_t)G3 * HID * 2);
  _Float16* wEncWih1  = (_Float16*)alloc((size_t)G3 * HID * 2);
  _Float16* wEncWhh1  = (_Float16*)alloc((size_t)G3 * HID * 2);
  _Float16* wDecWhh0f = (_Float16*)alloc((size_t)G3 * HID * 2);
  _Float16* wDecWhh0b = (_Float16*)alloc((size_t)G3 * HID * 2);
  _Float16* wDecWih1f = (_Float16*)alloc((size_t)G3 * 512 * 2);
  _Float16* wDecWih1b = (_Float16*)alloc((size_t)G3 * 512 * 2);
  _Float16* wDecWhh1f = (_Float16*)alloc((size_t)G3 * HID * 2);
  _Float16* wDecWhh1b = (_Float16*)alloc((size_t)G3 * HID * 2);
  float* Ggi   = (float*)alloc((size_t)BATCH * G3 * 4);
  float* Ggh   = (float*)alloc((size_t)BATCH * G3 * 4);
  float* xt    = (float*)alloc((size_t)BATCH * 4);
  float* y_seq = (float*)alloc((size_t)TT * BATCH * 4);
  _Float16* s1 = (_Float16*)alloc((size_t)TT * BATCH * 512 * 2);
  float* wp = (float*)alloc(TT * 4);
  float* wf = (float*)alloc(TT * 4);
  float* wb = (float*)alloc(TT * 4);

  // --- prep: f16 weight copies (original layout), merge weights, zero state ---
  convert_f16<<<dim3((G3 * HID) / 256), 256, 0, stream>>>(enc_Whh0, wEncWhh0, G3 * HID);
  convert_f16<<<dim3((G3 * HID) / 256), 256, 0, stream>>>(enc_Wih1, wEncWih1, G3 * HID);
  convert_f16<<<dim3((G3 * HID) / 256), 256, 0, stream>>>(enc_Whh1, wEncWhh1, G3 * HID);
  convert_f16<<<dim3((G3 * HID) / 256), 256, 0, stream>>>(dWhh0f, wDecWhh0f, G3 * HID);
  convert_f16<<<dim3((G3 * HID) / 256), 256, 0, stream>>>(dWhh0b, wDecWhh0b, G3 * HID);
  convert_f16<<<dim3((G3 * 512) / 256), 256, 0, stream>>>(dWih1f, wDecWih1f, G3 * 512);
  convert_f16<<<dim3((G3 * 512) / 256), 256, 0, stream>>>(dWih1b, wDecWih1b, G3 * 512);
  convert_f16<<<dim3((G3 * HID) / 256), 256, 0, stream>>>(dWhh1f, wDecWhh1f, G3 * HID);
  convert_f16<<<dim3((G3 * HID) / 256), 256, 0, stream>>>(dWhh1b, wDecWhh1b, G3 * HID);
  prep_small<<<1, 32, 0, stream>>>(weight_power, weight_merge, wp, wf, wb);
  zero_f32<<<2048, 256, 0, stream>>>((float*)zstart, zcount);

  dim3 gg(BATCH / 128, G3 / 64);
  const int cb = (BATCH * HID) / 256;

  // --- encoder (sequential over T, batchnorm feedback per step) ---
  for (int t = 0; t < TT; ++t) {
    gemm_kernel<HID><<<gg, 256, 0, stream>>>(h0_16, wEncWhh0, Ggh);
    enc_combine0<<<cb, 256, 0, stream>>>(bbits, zfb, enc_Wih0, enc_bih0, enc_bhh0, Ggh, h0, h0_16);
    gemm_kernel<HID><<<gg, 256, 0, stream>>>(h0_16, wEncWih1, Ggi);
    gemm_kernel<HID><<<gg, 256, 0, stream>>>(h1_16, wEncWhh1, Ggh);
    combine2<<<cb, 256, 0, stream>>>(Ggi, Ggh, enc_bih1, enc_bhh1, h1, h1_16);
    enc_rowreduce<<<(BATCH * 32) / 256, 256, 0, stream>>>(h1, enc_lin_W, enc_lin_b, xt, stats + 2 * t);
    enc_emit<<<BATCH / 256, 256, 0, stream>>>(xt, stats + 2 * t, wp, t, noise1, noise2, y_seq, zfb);
  }

  // --- decoder layer 0, forward and backward ---
  for (int t = 0; t < TT; ++t) {
    gemm_kernel<HID><<<gg, 256, 0, stream>>>(hd0f16, wDecWhh0f, Ggh);
    dec0_combine<<<cb, 256, 0, stream>>>(y_seq + (size_t)t * BATCH, dWih0f, dbih0f, dbhh0f,
                                         Ggh, hd0f, hd0f16, s1 + (size_t)t * BATCH * 512);
  }
  for (int t = TT - 1; t >= 0; --t) {
    gemm_kernel<HID><<<gg, 256, 0, stream>>>(hd0b16, wDecWhh0b, Ggh);
    dec0_combine<<<cb, 256, 0, stream>>>(y_seq + (size_t)t * BATCH, dWih0b, dbih0b, dbhh0b,
                                         Ggh, hd0b, hd0b16, s1 + (size_t)t * BATCH * 512 + HID);
  }

  // --- decoder layer 1, forward and backward, fused time-merge ---
  for (int t = 0; t < TT; ++t) {
    gemm_kernel<512><<<gg, 256, 0, stream>>>(s1 + (size_t)t * BATCH * 512, wDecWih1f, Ggi);
    gemm_kernel<HID><<<gg, 256, 0, stream>>>(hd1f16, wDecWhh1f, Ggh);
    dec1_combine<<<cb, 256, 0, stream>>>(Ggi, Ggh, dbih1f, dbhh1f, hd1f, hd1f16, rf, wf, t);
  }
  for (int t = TT - 1; t >= 0; --t) {
    gemm_kernel<512><<<gg, 256, 0, stream>>>(s1 + (size_t)t * BATCH * 512, wDecWih1b, Ggi);
    gemm_kernel<HID><<<gg, 256, 0, stream>>>(hd1b16, wDecWhh1b, Ggh);
    dec1_combine<<<cb, 256, 0, stream>>>(Ggi, Ggh, dbih1b, dbhh1b, hd1b, hd1b16, rb, wb, t);
  }

  // --- output head ---
  final_out<<<(BATCH * NBITS) / 256, 256, 0, stream>>>(rf, rb, dec_lin_W, dec_lin_b, out);
}
